// NeuralSDE_68066641707618
// MI455X (gfx1250) — matrix operand order repaired and checked by reading, not compiled, not run on hardware
//
#include <hip/hip_runtime.h>
#include <math.h>

#define B_   1024
#define T_   100
#define IN_  64
#define H_   512
#define OUT_ 64

typedef __bf16 bf16;
typedef bf16  v16bf __attribute__((ext_vector_type(16)));
typedef float v8f   __attribute__((ext_vector_type(8)));
typedef int   i32x4 __attribute__((ext_vector_type(4)));

union Frag16 { v16bf v; i32x4 q[2]; };

__device__ __forceinline__ unsigned short f2bf(float f) {
  union { float f; unsigned u; } x; x.f = f;
  unsigned r = x.u + 0x7FFFu + ((x.u >> 16) & 1u);   // round-to-nearest-even
  return (unsigned short)(r >> 16);
}

__device__ __forceinline__ v8f vzero() {
  v8f r = {0.f,0.f,0.f,0.f,0.f,0.f,0.f,0.f};
  return r;
}

__device__ __forceinline__ v8f wmma_bf16(const Frag16& a, const Frag16& b, v8f c) {
  return __builtin_amdgcn_wmma_f32_16x16x32_bf16(false, a.v, false, b.v,
                                                 (short)0, c, false, false);
}

// ---------------------------------------------------------------------------
// Pack a K x N row-major fp32 weight into WMMA-B-fragment-native bf16 layout:
// dst[((nt*KT + kt)*32 + lane)*16 + j]  holds element
//   (K = kt*32 + (lane/16)*16 + j,  N = nt*16 + lane%16)
// so a B fragment is 32 contiguous bytes per lane.
// ---------------------------------------------------------------------------
__global__ void pack_w(const float* __restrict__ src,
                       unsigned short* __restrict__ dst, int K, int N) {
  int i = blockIdx.x * blockDim.x + threadIdx.x;
  int total = K * N;
  if (i >= total) return;
  int KT   = K >> 5;
  int j    = i & 15;
  int lane = (i >> 4) & 31;
  int rem  = i >> 9;
  int kt   = rem % KT;
  int nt   = rem / KT;
  int k = (kt << 5) + ((lane >> 4) << 4) + j;
  int n = (nt << 4) + (lane & 15);
  dst[i] = f2bf(src[(size_t)k * N + n]);
}

// ---------------------------------------------------------------------------
// Wave-level GEMM: this wave accumulates NT 16x16 output tiles
//   rows  = arowBase + 0..15 of the LDS A buffer (bf16, pitch ldp halves)
//   ntile = ntBase..ntBase+NT-1 of the packed weight (K = KT*32)
// ---------------------------------------------------------------------------
template<int NT>
__device__ __forceinline__ void gemmN(const unsigned short* abuf, int ldp,
                                      const unsigned short* __restrict__ wpk,
                                      int KT, int ntBase, int lane, int arowBase,
                                      v8f* acc) {
  const int arow = arowBase + (lane & 15);
  const int kg   = (lane >> 4) << 4;                  // 0 or 16
  for (int kt = 0; kt < KT; ++kt) {
    Frag16 a;
    const unsigned short* ap = abuf + arow * ldp + (kt << 5) + kg;
    a.q[0] = *(const i32x4*)(ap);                     // K = kt*32 + kg + 0..7
    a.q[1] = *(const i32x4*)(ap + 8);                 // K = kt*32 + kg + 8..15
#pragma unroll
    for (int nt = 0; nt < NT; ++nt) {
      Frag16 b;
      const unsigned short* bp =
          wpk + (size_t)((((ntBase + nt) * KT) + kt) * 32 + lane) * 16;
      b.q[0] = *(const i32x4*)(bp);
      b.q[1] = *(const i32x4*)(bp + 8);
      acc[nt] = wmma_bf16(a, b, acc[nt]);
    }
  }
}

// ---------------------------------------------------------------------------
// SDE scan: one workgroup owns 16 batch rows for all 99 steps.
// 8 waves; wave w computes output columns [64w, 64w+64) (4 n-tiles).
// y (fp32) lives in registers of the owning wave; bf16 shadow in LDS is the
// A operand for the 4 per-step GEMMs. Scratch buffer holds tanh(...) interms.
// ---------------------------------------------------------------------------
__global__ void __launch_bounds__(256, 1) sde_scan(
    const float* __restrict__ coeffs, const float* __restrict__ times,
    const float* __restrict__ dW,
    const unsigned short* __restrict__ pWinit, const float* __restrict__ binit,
    const unsigned short* __restrict__ pWf1,   const float* __restrict__ bf1,
    const unsigned short* __restrict__ pWf2,   const float* __restrict__ bf2,
    const unsigned short* __restrict__ pWg1,   const float* __restrict__ bg1,
    const unsigned short* __restrict__ pWg2,   const float* __restrict__ bg2,
    float* __restrict__ zout) {
  const int LDP = H_ + 8;                             // 520 halves: 16B-aligned rows
  __shared__ __attribute__((aligned(16))) unsigned short ybuf[16 * (H_ + 8)];
  __shared__ __attribute__((aligned(16))) unsigned short sbuf[16 * (H_ + 8)];

  const int tid  = threadIdx.x;
  const int lane = tid & 31;
  const int w    = tid >> 5;
  const int b0   = blockIdx.x * 16;
  const int ntBase = w * 4;                           // wave's first n-tile
  const int mrow = (lane >> 4) << 3;                  // C-frag row base (+v)
  const int ncol = lane & 15;                         // C-frag col within tile
  const int KT   = H_ >> 5;                           // 16

  // stage x0 = coeffs[:,0,0,:] as bf16 A operand
  for (int i = tid; i < 16 * IN_; i += 256) {
    int m = i >> 6, c = i & 63;
    sbuf[m * LDP + c] =
        f2bf(coeffs[(size_t)(b0 + m) * ((T_ - 1) * 4 * IN_) + c]);
  }
  __syncthreads();

  // y0 = x0 @ W_init + b_init
  float yreg[4][8];
  {
    v8f acc[4];
#pragma unroll
    for (int nt = 0; nt < 4; ++nt) acc[nt] = vzero();
    gemmN<4>(sbuf, LDP, pWinit, IN_ >> 5, ntBase, lane, 0, acc);
#pragma unroll
    for (int nt = 0; nt < 4; ++nt) {
      int n = ((ntBase + nt) << 4) + ncol;
      float bb = binit[n];
#pragma unroll
      for (int v = 0; v < 8; ++v) {
        float y = acc[nt][v] + bb;
        yreg[nt][v] = y;
        ybuf[(mrow + v) * LDP + n] = f2bf(y);
        zout[(size_t)(b0 + mrow + v) * (T_ * H_) + n] = y;   // z[:,0,:]
      }
    }
  }
  __syncthreads();

  for (int t = 0; t < T_ - 1; ++t) {
    float h  = times[t + 1] - times[t];
    float sh = sqrtf(h);

    // u = tanh(y @ Wf1 + bf1) -> sbuf
    v8f acc[4];
#pragma unroll
    for (int nt = 0; nt < 4; ++nt) acc[nt] = vzero();
    gemmN<4>(ybuf, LDP, pWf1, KT, ntBase, lane, 0, acc);
#pragma unroll
    for (int nt = 0; nt < 4; ++nt) {
      int n = ((ntBase + nt) << 4) + ncol;
      float bb = bf1[n];
#pragma unroll
      for (int v = 0; v < 8; ++v)
        sbuf[(mrow + v) * LDP + n] = f2bf(tanhf(acc[nt][v] + bb));
    }
    __syncthreads();

    // facc = u @ Wf2   (bias added at update)
    v8f facc[4];
#pragma unroll
    for (int nt = 0; nt < 4; ++nt) facc[nt] = vzero();
    gemmN<4>(sbuf, LDP, pWf2, KT, ntBase, lane, 0, facc);
    __syncthreads();                                  // sbuf reads done

    // p = tanh(y @ Wg1 + bg1) -> sbuf
#pragma unroll
    for (int nt = 0; nt < 4; ++nt) acc[nt] = vzero();
    gemmN<4>(ybuf, LDP, pWg1, KT, ntBase, lane, 0, acc);
#pragma unroll
    for (int nt = 0; nt < 4; ++nt) {
      int n = ((ntBase + nt) << 4) + ncol;
      float bb = bg1[n];
#pragma unroll
      for (int v = 0; v < 8; ++v)
        sbuf[(mrow + v) * LDP + n] = f2bf(tanhf(acc[nt][v] + bb));
    }
    __syncthreads();                                  // ybuf reads also done

    // g = tanh(p @ Wg2 + bg2) ; y += f*h + g*sqrt(h)*dW[t]
#pragma unroll
    for (int nt = 0; nt < 4; ++nt) acc[nt] = vzero();
    gemmN<4>(sbuf, LDP, pWg2, KT, ntBase, lane, 0, acc);

    const float* dwt = dW + (size_t)t * B_ * H_;
#pragma unroll
    for (int nt = 0; nt < 4; ++nt) {
      int n = ((ntBase + nt) << 4) + ncol;
      float b2 = bf2[n];
      float bg = bg2[n];
#pragma unroll
      for (int v = 0; v < 8; ++v) {
        int   gm = b0 + mrow + v;
        float dw = dwt[(size_t)gm * H_ + n];
        float g  = tanhf(acc[nt][v] + bg);
        float yn = yreg[nt][v] + (facc[nt][v] + b2) * h + g * sh * dw;
        yreg[nt][v] = yn;
        ybuf[(mrow + v) * LDP + n] = f2bf(yn);
        zout[(size_t)gm * (T_ * H_) + (size_t)(t + 1) * H_ + n] = yn;
      }
    }
    __syncthreads();
  }
}

// ---------------------------------------------------------------------------
// Readout: out = relu(tanh(z) @ Wl1 + bl1) @ Wl2 + bl2
// One workgroup = 32 rows of the 102400 x 512 activation matrix.
// Wave w: M-tile (w&1), 8 n-tiles at base (w>>1)*8 for GEMM1;
//         job (mt=w&1, nt=w>>1) for GEMM2. Single LDS buffer reused.
// ---------------------------------------------------------------------------
__global__ void __launch_bounds__(256, 1) readout(
    const float* __restrict__ z,
    const unsigned short* __restrict__ pWl1, const float* __restrict__ bl1,
    const unsigned short* __restrict__ pWl2, const float* __restrict__ bl2,
    float* __restrict__ out) {
  const int LDP = H_ + 8;
  __shared__ __attribute__((aligned(16))) unsigned short abuf[32 * (H_ + 8)];

  const int tid  = threadIdx.x;
  const int lane = tid & 31;
  const int w    = tid >> 5;
  const int r0   = blockIdx.x * 32;
  const int mt   = w & 1;
  const int ntBase = (w >> 1) * 8;
  const int mrow = (lane >> 4) << 3;
  const int ncol = lane & 15;
  const int KT   = H_ >> 5;

  // a = tanh(z tile) as bf16
  for (int i = tid; i < 32 * H_; i += 256) {
    int m = i >> 9, c = i & (H_ - 1);
    abuf[m * LDP + c] = f2bf(tanhf(z[(size_t)(r0 + m) * H_ + c]));
  }
  __syncthreads();

  // h = relu(a @ Wl1 + bl1), kept in registers
  v8f acc[8];
#pragma unroll
  for (int nt = 0; nt < 8; ++nt) acc[nt] = vzero();
  gemmN<8>(abuf, LDP, pWl1, KT, ntBase, lane, mt * 16, acc);
#pragma unroll
  for (int nt = 0; nt < 8; ++nt) {
    int n = ((ntBase + nt) << 4) + ncol;
    float bb = bl1[n];
#pragma unroll
    for (int v = 0; v < 8; ++v) {
      float x = acc[nt][v] + bb;
      acc[nt][v] = x > 0.f ? x : 0.f;
    }
  }
  __syncthreads();                                    // all abuf reads done
#pragma unroll
  for (int nt = 0; nt < 8; ++nt) {
    int n = ((ntBase + nt) << 4) + ncol;
#pragma unroll
    for (int v = 0; v < 8; ++v)
      abuf[(mt * 16 + mrow + v) * LDP + n] = f2bf(acc[nt][v]);
  }
  __syncthreads();

  // out tile = h @ Wl2 + bl2
  v8f o[1];
  o[0] = vzero();
  int nt2 = w >> 1;                                   // 0..3 (OUT = 4 tiles)
  gemmN<1>(abuf, LDP, pWl2, KT, nt2, lane, mt * 16, o);
#pragma unroll
  for (int v = 0; v < 8; ++v) {
    int r = r0 + mt * 16 + mrow + v;
    int c = (nt2 << 4) + ncol;
    out[(size_t)r * OUT_ + c] = o[0][v] + bl2[c];
  }
}

// ---------------------------------------------------------------------------
extern "C" void kernel_launch(void* const* d_in, const int* in_sizes, int n_in,
                              void* d_out, int out_size, void* d_ws, size_t ws_size,
                              hipStream_t stream) {
  const float* coeffs = (const float*)d_in[0];
  const float* times  = (const float*)d_in[1];
  const float* dW     = (const float*)d_in[2];
  const float* Winit  = (const float*)d_in[3];
  const float* binit  = (const float*)d_in[4];
  const float* Wf1 = (const float*)d_in[5];  const float* bf1 = (const float*)d_in[6];
  const float* Wf2 = (const float*)d_in[7];  const float* bf2 = (const float*)d_in[8];
  const float* Wg1 = (const float*)d_in[9];  const float* bg1 = (const float*)d_in[10];
  const float* Wg2 = (const float*)d_in[11]; const float* bg2 = (const float*)d_in[12];
  const float* Wl1 = (const float*)d_in[13]; const float* bl1 = (const float*)d_in[14];
  const float* Wl2 = (const float*)d_in[15]; const float* bl2 = (const float*)d_in[16];

  float* out = (float*)d_out;
  float* z   = out + (size_t)B_ * T_ * OUT_;          // z follows out in d_out

  unsigned short* ws   = (unsigned short*)d_ws;
  unsigned short* pWf1 = ws;                 ws += (size_t)H_ * H_;
  unsigned short* pWf2 = ws;                 ws += (size_t)H_ * H_;
  unsigned short* pWg1 = ws;                 ws += (size_t)H_ * H_;
  unsigned short* pWg2 = ws;                 ws += (size_t)H_ * H_;
  unsigned short* pWl1 = ws;                 ws += (size_t)H_ * H_;
  unsigned short* pWl2 = ws;                 ws += (size_t)H_ * OUT_;
  unsigned short* pWinit = ws;

  const int HH = H_ * H_;
  pack_w<<<(HH + 255) / 256, 256, 0, stream>>>(Wf1, pWf1, H_, H_);
  pack_w<<<(HH + 255) / 256, 256, 0, stream>>>(Wf2, pWf2, H_, H_);
  pack_w<<<(HH + 255) / 256, 256, 0, stream>>>(Wg1, pWg1, H_, H_);
  pack_w<<<(HH + 255) / 256, 256, 0, stream>>>(Wg2, pWg2, H_, H_);
  pack_w<<<(HH + 255) / 256, 256, 0, stream>>>(Wl1, pWl1, H_, H_);
  pack_w<<<(H_ * OUT_ + 255) / 256, 256, 0, stream>>>(Wl2, pWl2, H_, OUT_);
  pack_w<<<(IN_ * H_ + 255) / 256, 256, 0, stream>>>(Winit, pWinit, IN_, H_);

  sde_scan<<<B_ / 16, 256, 0, stream>>>(coeffs, times, dW,
                                        pWinit, binit, pWf1, bf1, pWf2, bf2,
                                        pWg1, bg1, pWg2, bg2, z);
  readout<<<(B_ * T_) / 32, 256, 0, stream>>>(z, pWl1, bl1, pWl2, bl2, out);
}
